// GraphEncoder_36782099923301
// MI455X (gfx1250) — matrix operand order repaired and checked by reading, not compiled, not run on hardware
//
#include <hip/hip_runtime.h>
#include <hip/hip_bf16.h>
#include <cstdint>

// ---------------------------------------------------------------------------
// Types for CDNA5 WMMA
// ---------------------------------------------------------------------------
typedef __attribute__((ext_vector_type(16))) __bf16 v16bf;
typedef __attribute__((ext_vector_type(8)))  float  v8f;

union Frag16 { v16bf v; unsigned u[8]; };

#define NEG_SLOPE 0.2f

// ---------------------------------------------------------------------------
// Helpers
// ---------------------------------------------------------------------------
__device__ __forceinline__ unsigned short f2bf(float f) {
  unsigned u = __float_as_uint(f);
  u += 0x7FFFu + ((u >> 16) & 1u);   // round-to-nearest-even
  return (unsigned short)(u >> 16);
}

// order-preserving float <-> uint mapping (for atomic max on floats)
__device__ __forceinline__ unsigned f2ord(float f) {
  unsigned u = __float_as_uint(f);
  return (u & 0x80000000u) ? ~u : (u | 0x80000000u);
}
__device__ __forceinline__ float ord2f(unsigned u) {
  return __uint_as_float((u & 0x80000000u) ? (u & 0x7FFFFFFFu) : ~u);
}

__device__ __forceinline__ void edge_sd(const int* __restrict__ ei, int E, int e,
                                        int& s, int& d) {
  if (e < E) { s = ei[e]; d = ei[E + e]; }
  else       { int l = e - E; s = l; d = l; }   // self loops appended
}

// CDNA5 async Global->LDS copy (16B per lane), tracked by ASYNCcnt.
__device__ __forceinline__ void async_b128_to_lds(unsigned lds_off, const void* gptr) {
  asm volatile("global_load_async_to_lds_b128 %0, %1, off"
               :: "v"(lds_off), "v"(gptr)
               : "memory");
}

// ---------------------------------------------------------------------------
// Elementwise kernels
// ---------------------------------------------------------------------------
__global__ __launch_bounds__(256) void k_f32_to_bf16(const float* __restrict__ in,
                                                     unsigned short* __restrict__ out,
                                                     int n) {
  int i = blockIdx.x * 256 + threadIdx.x;
  if (i < n) out[i] = f2bf(in[i]);
}

// W[k][n] (f32, KxN row-major) -> Wt[n][k] (bf16, NxK row-major)
__global__ __launch_bounds__(256) void k_transpose_to_bf16(const float* __restrict__ W,
                                                           unsigned short* __restrict__ Wt,
                                                           int K, int N) {
  int i = blockIdx.x * 256 + threadIdx.x;
  if (i >= K * N) return;
  int n = i / K, k = i - n * K;
  Wt[i] = f2bf(W[(size_t)k * N + n]);
}

__global__ __launch_bounds__(256) void k_fill_f32(float* p, float v, int n) {
  int i = blockIdx.x * 256 + threadIdx.x;
  if (i < n) p[i] = v;
}
__global__ __launch_bounds__(256) void k_fill_u32(unsigned* p, unsigned v, int n) {
  int i = blockIdx.x * 256 + threadIdx.x;
  if (i < n) p[i] = v;
}

// ---------------------------------------------------------------------------
// bf16 GEMM with WMMA: C[M,N] = A[M,K] * Bt[N,K]^T  (f32 accumulate)
// 128x128x32 block tile, 8 waves (2 along M x 4 along N), wave tile 64x32.
// Double-buffered LDS filled by async global->LDS copies (ASYNCcnt pipeline).
// ---------------------------------------------------------------------------
#define BM 128
#define BN 128
#define BK 32
#define LDP 40   // padded LDS row (80B: 16B-aligned, bank-spread)
#define TILE_BYTES (BM * LDP * 2)

// issue one 128x32 A tile + 128x32 B tile (4 x b128 per thread) asynchronously
__device__ __forceinline__ void tile_issue(const unsigned short* __restrict__ A,
                                           const unsigned short* __restrict__ Bt,
                                           int M, int K, int bm0, int bn0, int k0,
                                           unsigned ldsA, unsigned ldsB, int tid) {
  #pragma unroll
  for (int i = 0; i < 2; ++i) {
    int idx = tid + i * 256;
    int r = idx >> 2;
    int c = (idx & 3) * 8;
    int gm = bm0 + r;
    if (gm >= M) gm = M - 1;   // clamp: padded rows get duplicate (harmless) data
    async_b128_to_lds(ldsA + (unsigned)(r * LDP + c) * 2u,
                      A + (size_t)gm * K + k0 + c);
    async_b128_to_lds(ldsB + (unsigned)(r * LDP + c) * 2u,
                      Bt + (size_t)(bn0 + r) * K + k0 + c);
  }
}

__global__ __launch_bounds__(256) void k_gemm_bf16(
    const unsigned short* __restrict__ A,   // [M,K] bf16
    const unsigned short* __restrict__ Bt,  // [N,K] bf16
    float* __restrict__ C, int M, int N, int K) {
  __shared__ __align__(16) unsigned short lA[2][BM][LDP];
  __shared__ __align__(16) unsigned short lB[2][BN][LDP];

  const int tid  = threadIdx.x;
  const int lane = tid & 31;
  const int w    = tid >> 5;
  const int wm   = w & 1;        // 0..1 over M
  const int wn   = w >> 1;       // 0..3 over N
  const int ml   = lane & 15;
  const int half = lane >> 4;
  const int bm0  = blockIdx.y * BM;
  const int bn0  = blockIdx.x * BN;

  const unsigned baseA = (unsigned)(uintptr_t)&lA[0][0][0];
  const unsigned baseB = (unsigned)(uintptr_t)&lB[0][0][0];

  v8f acc[4][2];
  #pragma unroll
  for (int s = 0; s < 4; ++s)
    #pragma unroll
    for (int t = 0; t < 2; ++t)
      #pragma unroll
      for (int j = 0; j < 8; ++j) acc[s][t][j] = 0.0f;

  const int nk = K / BK;
  // prologue: fill buffer 0
  tile_issue(A, Bt, M, K, bm0, bn0, 0, baseA, baseB, tid);

  for (int kt = 0; kt < nk; ++kt) {
    const int cur = kt & 1;
    if (kt + 1 < nk) {
      // overlap: start next tile into the other buffer, then wait for current
      tile_issue(A, Bt, M, K, bm0, bn0, (kt + 1) * BK,
                 baseA + (unsigned)((cur ^ 1) * TILE_BYTES),
                 baseB + (unsigned)((cur ^ 1) * TILE_BYTES), tid);
      asm volatile("s_wait_asynccnt 4" ::: "memory");   // current tile landed
    } else {
      asm volatile("s_wait_asynccnt 0" ::: "memory");
    }
    __syncthreads();

    // gather fragments: contiguous K order (lane half h holds k = 16h..16h+15)
    Frag16 af[4], bf[2];
    #pragma unroll
    for (int s = 0; s < 4; ++s) {
      const unsigned short* row = &lA[cur][wm * 64 + s * 16 + ml][0];
      #pragma unroll
      for (int v = 0; v < 8; ++v) {
        int k = 2 * v + half * 16;
        af[s].u[v] = *(const unsigned*)(row + k);
      }
    }
    #pragma unroll
    for (int t = 0; t < 2; ++t) {
      const unsigned short* row = &lB[cur][wn * 32 + t * 16 + ml][0];
      #pragma unroll
      for (int v = 0; v < 8; ++v) {
        int k = 2 * v + half * 16;
        bf[t].u[v] = *(const unsigned*)(row + k);
      }
    }

    #pragma unroll
    for (int s = 0; s < 4; ++s)
      #pragma unroll
      for (int t = 0; t < 2; ++t)
        acc[s][t] = __builtin_amdgcn_wmma_f32_16x16x32_bf16(
            false, af[s].v, false, bf[t].v, (short)0, acc[s][t], false, false);

    __syncthreads();   // all reads done before next iteration overwrites buffer
  }

  // store: VGPR j -> M = j + 8*half (ISA C/D layout), N = lane%16
  #pragma unroll
  for (int s = 0; s < 4; ++s)
    #pragma unroll
    for (int t = 0; t < 2; ++t) {
      int n = bn0 + wn * 32 + t * 16 + ml;
      #pragma unroll
      for (int j = 0; j < 8; ++j) {
        int m = bm0 + wm * 64 + s * 16 + half * 8 + j;
        if (m < M) C[(size_t)m * N + n] = acc[s][t][j];
      }
    }
}

// ---------------------------------------------------------------------------
// attention scores: als/ald[n,h] = <h[n,h,:], a_src/a_dst[h,:]> (wave per row)
// ---------------------------------------------------------------------------
__global__ __launch_bounds__(256) void k_att(const float* __restrict__ h,
                                             const float* __restrict__ as,
                                             const float* __restrict__ ad,
                                             float* __restrict__ als,
                                             float* __restrict__ ald,
                                             int Nn, int H, int C) {
  int w = blockIdx.x * 8 + (threadIdx.x >> 5);
  int lane = threadIdx.x & 31;
  if (w >= Nn * H) return;
  int n = w / H, hd = w - n * H;
  const float* hp  = h + (size_t)n * H * C + (size_t)hd * C;
  const float* asp = as + hd * C;
  const float* adp = ad + hd * C;
  float s0 = 0.f, s1 = 0.f;
  for (int c = lane; c < C; c += 32) {
    float v = hp[c];
    s0 += v * asp[c];
    s1 += v * adp[c];
  }
  #pragma unroll
  for (int off = 16; off; off >>= 1) {
    s0 += __shfl_xor(s0, off, 32);
    s1 += __shfl_xor(s1, off, 32);
  }
  if (lane == 0) { als[w] = s0; ald[w] = s1; }
}

// ---------------------------------------------------------------------------
// edge passes
// ---------------------------------------------------------------------------
__global__ __launch_bounds__(256) void k_edge_max(const int* __restrict__ ei, int E, int Etot,
                                                  const float* __restrict__ als,
                                                  const float* __restrict__ ald,
                                                  unsigned* __restrict__ m,
                                                  float* __restrict__ ev, int H) {
  int i = blockIdx.x * 256 + threadIdx.x;
  if (i >= Etot * H) return;
  int e = i / H, hd = i - e * H;
  int s, d; edge_sd(ei, E, e, s, d);
  float v = als[s * H + hd] + ald[d * H + hd];
  v = v > 0.f ? v : v * NEG_SLOPE;
  ev[i] = v;
  atomicMax(&m[d * H + hd], f2ord(v));
}

__global__ __launch_bounds__(256) void k_edge_den(const int* __restrict__ ei, int E, int Etot,
                                                  const float* __restrict__ ev,
                                                  const unsigned* __restrict__ m,
                                                  float* __restrict__ den, int H) {
  int i = blockIdx.x * 256 + threadIdx.x;
  if (i >= Etot * H) return;
  int e = i / H, hd = i - e * H;
  int s, d; edge_sd(ei, E, e, s, d);
  (void)s;
  float ex = __expf(ev[i] - ord2f(m[d * H + hd]));
  atomicAdd(&den[d * H + hd], ex);
}

// block per edge: agg[dst] += h[src] * alpha  (L2-resident scatter, f32 atomics)
__global__ __launch_bounds__(256) void k_edge_agg(const int* __restrict__ ei, int E,
                                                  const float* __restrict__ ev,
                                                  const unsigned* __restrict__ m,
                                                  const float* __restrict__ den,
                                                  const float* __restrict__ h,
                                                  float* __restrict__ agg,
                                                  int H, int C) {
  __shared__ float s_alpha[8];
  int e = blockIdx.x;
  int s, d; edge_sd(ei, E, e, s, d);
  if ((int)threadIdx.x < H) {
    int hd = threadIdx.x;
    float ex = __expf(ev[e * H + hd] - ord2f(m[d * H + hd]));
    s_alpha[hd] = ex / (den[d * H + hd] + 1e-16f);
  }
  __syncthreads();
  int F = H * C;
  const float* hs = h + (size_t)s * F;
  float* ag = agg + (size_t)d * F;
  for (int i = threadIdx.x; i < F; i += 256)
    atomicAdd(&ag[i], hs[i] * s_alpha[i / C]);
}

// ---------------------------------------------------------------------------
// epilogues
// ---------------------------------------------------------------------------
__global__ __launch_bounds__(256) void k_bias_elu_bf16(const float* __restrict__ agg,
                                                       const float* __restrict__ b,
                                                       unsigned short* __restrict__ out,
                                                       int total, int F) {
  int i = blockIdx.x * 256 + threadIdx.x;
  if (i >= total) return;
  float v = agg[i] + b[i % F];
  v = v > 0.f ? v : (__expf(v) - 1.0f);   // ELU(alpha=1)
  out[i] = f2bf(v);
}

__global__ __launch_bounds__(256) void k_bias_out(const float* __restrict__ agg,
                                                  const float* __restrict__ b,
                                                  float* __restrict__ out,
                                                  int total, int F) {
  int i = blockIdx.x * 256 + threadIdx.x;
  if (i >= total) return;
  out[i] = agg[i] + b[i % F];
}

// ---------------------------------------------------------------------------
// launch
// ---------------------------------------------------------------------------
extern "C" void kernel_launch(void* const* d_in, const int* in_sizes, int n_in,
                              void* d_out, int out_size, void* d_ws, size_t ws_size,
                              hipStream_t stream) {
  (void)n_in; (void)out_size; (void)ws_size;
  const float* x   = (const float*)d_in[0];
  const int*   ei  = (const int*)d_in[1];
  const float* W1  = (const float*)d_in[2];
  const float* as1 = (const float*)d_in[3];
  const float* ad1 = (const float*)d_in[4];
  const float* b1  = (const float*)d_in[5];
  const float* W2  = (const float*)d_in[6];
  const float* as2 = (const float*)d_in[7];
  const float* ad2 = (const float*)d_in[8];
  const float* b2  = (const float*)d_in[9];
  float* out = (float*)d_out;

  const int IN = 768, H1 = 4, C1 = 512, F1 = 2048, C2 = 768;
  const int Nn   = in_sizes[0] / IN;   // 10000
  const int E    = in_sizes[1] / 2;    // 160000
  const int Etot = E + Nn;             // + self loops

  size_t cur = 0;
  auto carve = [&](size_t bytes) -> void* {
    void* p = (void*)((char*)d_ws + cur);
    cur += (bytes + 255) & ~(size_t)255;
    return p;
  };
  unsigned short* xb   = (unsigned short*)carve((size_t)Nn * IN * 2);
  unsigned short* w1t  = (unsigned short*)carve((size_t)F1 * IN * 2);
  float*          h1   = (float*)carve((size_t)Nn * F1 * 4);
  float*          als1 = (float*)carve((size_t)Nn * H1 * 4);
  float*          ald1 = (float*)carve((size_t)Nn * H1 * 4);
  unsigned*       m1   = (unsigned*)carve((size_t)Nn * H1 * 4);
  float*          den1 = (float*)carve((size_t)Nn * H1 * 4);
  float*          ev1  = (float*)carve((size_t)Etot * H1 * 4);
  float*          agg1 = (float*)carve((size_t)Nn * F1 * 4);
  unsigned short* heb  = (unsigned short*)carve((size_t)Nn * F1 * 2);
  unsigned short* w2t  = (unsigned short*)carve((size_t)C2 * F1 * 2);
  float*          h2   = (float*)carve((size_t)Nn * C2 * 4);
  float*          als2 = (float*)carve((size_t)Nn * 4);
  float*          ald2 = (float*)carve((size_t)Nn * 4);
  unsigned*       m2   = (unsigned*)carve((size_t)Nn * 4);
  float*          den2 = (float*)carve((size_t)Nn * 4);
  float*          ev2  = (float*)carve((size_t)Etot * 4);
  float*          agg2 = (float*)carve((size_t)Nn * C2 * 4);

  auto cdiv = [](int a, int b) { return (a + b - 1) / b; };

  // ---------------- layer 1 ----------------
  k_f32_to_bf16<<<cdiv(Nn * IN, 256), 256, 0, stream>>>(x, xb, Nn * IN);
  k_transpose_to_bf16<<<cdiv(IN * F1, 256), 256, 0, stream>>>(W1, w1t, IN, F1);
  {
    dim3 g(F1 / BN, cdiv(Nn, BM));
    k_gemm_bf16<<<g, 256, 0, stream>>>(xb, w1t, h1, Nn, F1, IN);
  }
  k_att<<<cdiv(Nn * H1, 8), 256, 0, stream>>>(h1, as1, ad1, als1, ald1, Nn, H1, C1);
  k_fill_u32<<<cdiv(Nn * H1, 256), 256, 0, stream>>>(m1, 0u, Nn * H1);
  k_fill_f32<<<cdiv(Nn * H1, 256), 256, 0, stream>>>(den1, 0.f, Nn * H1);
  k_fill_f32<<<cdiv(Nn * F1, 256), 256, 0, stream>>>(agg1, 0.f, Nn * F1);
  k_edge_max<<<cdiv(Etot * H1, 256), 256, 0, stream>>>(ei, E, Etot, als1, ald1, m1, ev1, H1);
  k_edge_den<<<cdiv(Etot * H1, 256), 256, 0, stream>>>(ei, E, Etot, ev1, m1, den1, H1);
  k_edge_agg<<<Etot, 256, 0, stream>>>(ei, E, ev1, m1, den1, h1, agg1, H1, C1);
  k_bias_elu_bf16<<<cdiv(Nn * F1, 256), 256, 0, stream>>>(agg1, b1, heb, Nn * F1, F1);

  // ---------------- layer 2 ----------------
  k_transpose_to_bf16<<<cdiv(F1 * C2, 256), 256, 0, stream>>>(W2, w2t, F1, C2);
  {
    dim3 g(C2 / BN, cdiv(Nn, BM));
    k_gemm_bf16<<<g, 256, 0, stream>>>(heb, w2t, h2, Nn, C2, F1);
  }
  k_att<<<cdiv(Nn, 8), 256, 0, stream>>>(h2, as2, ad2, als2, ald2, Nn, 1, C2);
  k_fill_u32<<<cdiv(Nn, 256), 256, 0, stream>>>(m2, 0u, Nn);
  k_fill_f32<<<cdiv(Nn, 256), 256, 0, stream>>>(den2, 0.f, Nn);
  k_fill_f32<<<cdiv(Nn * C2, 256), 256, 0, stream>>>(agg2, 0.f, Nn * C2);
  k_edge_max<<<cdiv(Etot, 256), 256, 0, stream>>>(ei, E, Etot, als2, ald2, m2, ev2, 1);
  k_edge_den<<<cdiv(Etot, 256), 256, 0, stream>>>(ei, E, Etot, ev2, m2, den2, 1);
  k_edge_agg<<<Etot, 256, 0, stream>>>(ei, E, ev2, m2, den2, h2, agg2, 1, C2);
  k_bias_out<<<cdiv(Nn * C2, 256), 256, 0, stream>>>(agg2, b2, out, Nn * C2, C2);
}
